// VisionTransformer_37641093382346
// MI455X (gfx1250) — compile-verified, matrix operand order checked by reading, not run on hardware
//
#include <hip/hip_runtime.h>
#include <hip/hip_bf16.h>

#define DEV __device__ __forceinline__

typedef __attribute__((ext_vector_type(16))) __bf16 v16bf;
typedef __attribute__((ext_vector_type(8)))  float  v8f;

struct __align__(16) B16v { unsigned long long lo, hi; };
union FragU { B16v b[2]; v16bf v; };
union HalfU { B16v b; __bf16 h[8]; };

// ---- model constants ----
static constexpr int Bsz    = 8;
static constexpr int CIN    = 3;
static constexpr int IMG    = 224;
static constexpr int Pp     = 16;
static constexpr int Dm     = 512;
static constexpr int Ll     = 8;
static constexpr int NHh    = 8;
static constexpr int Ee     = 8;
static constexpr int HIDn   = 2048;
static constexpr int NCLS   = 100;
static constexpr int GRIDp  = 13;
static constexpr int NPATCH = 169;          // 13*13
static constexpr int Ntok   = NPATCH + 1;   // 170
static constexpr int Ttok   = Bsz * Ntok;   // 1360
static constexpr int TP     = Bsz * NPATCH; // 1352
static constexpr int KPATCH = CIN * Pp * Pp;// 768
static constexpr float EPSf = 1.1920929e-07f;

DEV __bf16 f2bf(float f) {
  unsigned int u = __builtin_bit_cast(unsigned int, f);
  unsigned int r = (u + 0x7FFFu + ((u >> 16) & 1u)) >> 16;
  unsigned short s = (unsigned short)r;
  return __builtin_bit_cast(__bf16, s);
}
DEV float bf2f(__bf16 h) {
  unsigned short s = __builtin_bit_cast(unsigned short, h);
  unsigned int u = ((unsigned int)s) << 16;
  return __builtin_bit_cast(float, u);
}
DEV float gelu_f(float x) { return 0.5f * x * (1.f + erff(x * 0.70710678118654752f)); }

// Fragment layout (CDNA5 16-bit A 16x32): lane holds two contiguous 8-half chunks
// at [kb..kb+7] and [kb+16..kb+23]; kb = (lane>=16)?8:0.
DEV v16bf load_frag(const __bf16* p) {
  FragU u;
  u.b[0] = *(const B16v*)p;
  u.b[1] = *(const B16v*)(p + 16);
  return u.v;
}

DEV float block_sum_256(float p, float* red) {
  const int lane = threadIdx.x & 31, wave = threadIdx.x >> 5;
  #pragma unroll
  for (int off = 16; off; off >>= 1) p += __shfl_xor(p, off, 32);
  if (lane == 0) red[wave] = p;
  __syncthreads();
  float t = red[0];
  #pragma unroll
  for (int w = 1; w < 8; ++w) t += red[w];
  return t;
}

// =====================================================================
// Generic bf16 WMMA GEMM: C[M,N] = A[M,K]*B[K,N] + bias   (residual: C +=)
// Block: 64x64 tile, 8 waves (wave32), each wave -> 16x32 via two 16x16 acc.
// =====================================================================
__global__ __launch_bounds__(256) void gemm_bf16_kernel(
    const __bf16* __restrict__ A, const __bf16* __restrict__ Bw,
    const float* __restrict__ bias, float* __restrict__ C,
    int M, int N, int K, int residual)
{
  __shared__ __align__(16) __bf16 As[64][32];
  __shared__ __align__(16) __bf16 Bs[64][32];   // stored N-major (transposed tile)
  const int tid = threadIdx.x, lane = tid & 31, wave = tid >> 5;
  const int m0 = blockIdx.x * 64, n0 = blockIdx.y * 64;
  const int wm = (wave & 3) * 16, wn = (wave >> 2) * 32;
  v8f c0 = {0.f,0.f,0.f,0.f,0.f,0.f,0.f,0.f};
  v8f c1 = {0.f,0.f,0.f,0.f,0.f,0.f,0.f,0.f};
  const int ar = tid >> 2, ac = (tid & 3) * 8;  // A: 64 rows x 32 cols, 8 halves/thread
  const int br = tid >> 3, bc = (tid & 7) * 8;  // B: 32 rows x 64 cols, 8 halves/thread
  const int gm = m0 + ar;
  for (int k0 = 0; k0 < K; k0 += 32) {
    if (gm < M) {
      *(B16v*)&As[ar][ac] = *(const B16v*)(A + (size_t)gm * K + k0 + ac);
    } else {
      B16v z{0, 0};
      *(B16v*)&As[ar][ac] = z;
    }
    {
      HalfU tb;
      const int gn = n0 + bc;
      if (gn + 7 < N) tb.b = *(const B16v*)(Bw + (size_t)(k0 + br) * N + gn);
      else { tb.b.lo = 0; tb.b.hi = 0; }
      #pragma unroll
      for (int j = 0; j < 8; ++j) Bs[bc + j][br] = tb.h[j];
    }
    if (k0 + 32 < K) {
      if (gm < M) __builtin_prefetch((const void*)(A + (size_t)gm * K + k0 + 32 + ac), 0, 0);
      __builtin_prefetch((const void*)(Bw + (size_t)(k0 + 32 + br) * N + n0 + bc), 0, 0);
    }
    __syncthreads();
    const int kb = (lane >> 4) << 3;
    v16bf fa  = load_frag(&As[wm + (lane & 15)][kb]);
    v16bf fb0 = load_frag(&Bs[wn + (lane & 15)][kb]);
    v16bf fb1 = load_frag(&Bs[wn + 16 + (lane & 15)][kb]);
    c0 = __builtin_amdgcn_wmma_f32_16x16x32_bf16(false, fa, false, fb0, (short)0, c0, false, false);
    c1 = __builtin_amdgcn_wmma_f32_16x16x32_bf16(false, fa, false, fb1, (short)0, c1, false, false);
    __syncthreads();
  }
  const int col0 = n0 + wn + (lane & 15);
  const int rb = m0 + wm + ((lane >> 4) << 3);
  #pragma unroll
  for (int r = 0; r < 8; ++r) {
    const int row = rb + r;
    if (row < M && col0 + 16 < N) {
      const float v0 = c0[r] + bias[col0];
      const float v1 = c1[r] + bias[col0 + 16];
      const size_t i0 = (size_t)row * N + col0;
      if (residual) { C[i0] += v0; C[i0 + 16] += v1; }
      else          { C[i0]  = v0; C[i0 + 16]  = v1; }
    }
  }
}

// =====================================================================
// MoE expert GEMM 1 (gathered rows): Hid[e,slot,:] = gelu(Hn[tok]*W1[e] + b1[e])
// =====================================================================
__global__ __launch_bounds__(256) void moe_gemm1_kernel(
    const __bf16* __restrict__ Hn, const __bf16* __restrict__ W1,
    const float* __restrict__ B1, const int* __restrict__ counts,
    const int* __restrict__ tok, __bf16* __restrict__ Hid)
{
  const int e = blockIdx.z;
  const int cnt = counts[e];
  const int m0 = blockIdx.x * 64;
  if (m0 >= cnt) return;
  __shared__ __align__(16) __bf16 As[64][32];
  __shared__ __align__(16) __bf16 Bs[64][32];
  const int tid = threadIdx.x, lane = tid & 31, wave = tid >> 5;
  const int n0 = blockIdx.y * 64;
  const int wm = (wave & 3) * 16, wn = (wave >> 2) * 32;
  v8f c0 = {0.f,0.f,0.f,0.f,0.f,0.f,0.f,0.f};
  v8f c1 = {0.f,0.f,0.f,0.f,0.f,0.f,0.f,0.f};
  const int ar = tid >> 2, ac = (tid & 3) * 8;
  const int br = tid >> 3, bc = (tid & 7) * 8;
  int trow = -1;
  if (m0 + ar < cnt) trow = tok[e * Ttok + m0 + ar];
  const __bf16* Bw = W1 + (size_t)e * Dm * HIDn;
  for (int k0 = 0; k0 < Dm; k0 += 32) {
    if (trow >= 0) {
      *(B16v*)&As[ar][ac] = *(const B16v*)(Hn + (size_t)trow * Dm + k0 + ac);
    } else {
      B16v z{0, 0};
      *(B16v*)&As[ar][ac] = z;
    }
    {
      HalfU tb;
      tb.b = *(const B16v*)(Bw + (size_t)(k0 + br) * HIDn + n0 + bc);
      #pragma unroll
      for (int j = 0; j < 8; ++j) Bs[bc + j][br] = tb.h[j];
    }
    __syncthreads();
    const int kb = (lane >> 4) << 3;
    v16bf fa  = load_frag(&As[wm + (lane & 15)][kb]);
    v16bf fb0 = load_frag(&Bs[wn + (lane & 15)][kb]);
    v16bf fb1 = load_frag(&Bs[wn + 16 + (lane & 15)][kb]);
    c0 = __builtin_amdgcn_wmma_f32_16x16x32_bf16(false, fa, false, fb0, (short)0, c0, false, false);
    c1 = __builtin_amdgcn_wmma_f32_16x16x32_bf16(false, fa, false, fb1, (short)0, c1, false, false);
    __syncthreads();
  }
  const int col0 = n0 + wn + (lane & 15);
  const int rb = m0 + wm + ((lane >> 4) << 3);
  #pragma unroll
  for (int r = 0; r < 8; ++r) {
    const int row = rb + r;
    if (row < cnt) {
      const float v0 = gelu_f(c0[r] + B1[e * HIDn + col0]);
      const float v1 = gelu_f(c1[r] + B1[e * HIDn + col0 + 16]);
      const size_t i0 = ((size_t)e * Ttok + row) * HIDn + col0;
      Hid[i0] = f2bf(v0);
      Hid[i0 + 16] = f2bf(v1);
    }
  }
}

// =====================================================================
// MoE expert GEMM 2: Y[e,slot,:] = Hid[e,slot,:]*W2[e] + b2[e]
// =====================================================================
__global__ __launch_bounds__(256) void moe_gemm2_kernel(
    const __bf16* __restrict__ Hid, const __bf16* __restrict__ W2,
    const float* __restrict__ B2, const int* __restrict__ counts,
    float* __restrict__ Y)
{
  const int e = blockIdx.z;
  const int cnt = counts[e];
  const int m0 = blockIdx.x * 64;
  if (m0 >= cnt) return;
  __shared__ __align__(16) __bf16 As[64][32];
  __shared__ __align__(16) __bf16 Bs[64][32];
  const int tid = threadIdx.x, lane = tid & 31, wave = tid >> 5;
  const int n0 = blockIdx.y * 64;
  const int wm = (wave & 3) * 16, wn = (wave >> 2) * 32;
  v8f c0 = {0.f,0.f,0.f,0.f,0.f,0.f,0.f,0.f};
  v8f c1 = {0.f,0.f,0.f,0.f,0.f,0.f,0.f,0.f};
  const int ar = tid >> 2, ac = (tid & 3) * 8;
  const int br = tid >> 3, bc = (tid & 7) * 8;
  const bool arok = (m0 + ar < cnt);
  const __bf16* Arows = Hid + (size_t)e * Ttok * HIDn;
  const __bf16* Bw = W2 + (size_t)e * HIDn * Dm;
  for (int k0 = 0; k0 < HIDn; k0 += 32) {
    if (arok) {
      *(B16v*)&As[ar][ac] = *(const B16v*)(Arows + (size_t)(m0 + ar) * HIDn + k0 + ac);
    } else {
      B16v z{0, 0};
      *(B16v*)&As[ar][ac] = z;
    }
    {
      HalfU tb;
      tb.b = *(const B16v*)(Bw + (size_t)(k0 + br) * Dm + n0 + bc);
      #pragma unroll
      for (int j = 0; j < 8; ++j) Bs[bc + j][br] = tb.h[j];
    }
    __syncthreads();
    const int kb = (lane >> 4) << 3;
    v16bf fa  = load_frag(&As[wm + (lane & 15)][kb]);
    v16bf fb0 = load_frag(&Bs[wn + (lane & 15)][kb]);
    v16bf fb1 = load_frag(&Bs[wn + 16 + (lane & 15)][kb]);
    c0 = __builtin_amdgcn_wmma_f32_16x16x32_bf16(false, fa, false, fb0, (short)0, c0, false, false);
    c1 = __builtin_amdgcn_wmma_f32_16x16x32_bf16(false, fa, false, fb1, (short)0, c1, false, false);
    __syncthreads();
  }
  const int col0 = n0 + wn + (lane & 15);
  const int rb = m0 + wm + ((lane >> 4) << 3);
  #pragma unroll
  for (int r = 0; r < 8; ++r) {
    const int row = rb + r;
    if (row < cnt) {
      const size_t i0 = ((size_t)e * Ttok + row) * Dm + col0;
      Y[i0] = c0[r] + B2[e * Dm + col0];
      Y[i0 + 16] = c1[r] + B2[e * Dm + col0 + 16];
    }
  }
}

// =====================================================================
// im2col for dilated patch conv (bf16 output)
// =====================================================================
__global__ __launch_bounds__(256) void im2col_kernel(
    const float* __restrict__ X, __bf16* __restrict__ Ai)
{
  const size_t i = (size_t)blockIdx.x * 256 + threadIdx.x;
  if (i >= (size_t)TP * KPATCH) return;
  const int t = (int)(i / KPATCH), j = (int)(i % KPATCH);
  const int b = t / NPATCH, p = t % NPATCH;
  const int gy = p / GRIDp, gx = p % GRIDp;
  const int c = j / (Pp * Pp), rem = j % (Pp * Pp);
  const int ky = rem / Pp, kx = rem % Pp;
  const int iy = gy * 16 - 7 + ky * 2;
  const int ix = gx * 16 - 7 + kx * 2;
  float v = 0.f;
  if (iy >= 0 && iy < IMG && ix >= 0 && ix < IMG)
    v = X[(((size_t)b * CIN + c) * IMG + iy) * IMG + ix];
  Ai[i] = f2bf(v);
}

// patch tokens: rms + pos; cls token path for n==0
__global__ __launch_bounds__(256) void patch_post_kernel(
    const float* __restrict__ Pout, const float* __restrict__ pnw,
    const float* __restrict__ cls, const float* __restrict__ pos,
    float* __restrict__ Z)
{
  __shared__ float red[8];
  const int token = blockIdx.x;
  const int b = token / Ntok, n = token % Ntok;
  const int tid = threadIdx.x;
  if (n == 0) {
    Z[(size_t)token * Dm + tid]       = cls[tid]       + pos[tid];
    Z[(size_t)token * Dm + tid + 256] = cls[tid + 256] + pos[tid + 256];
    return;
  }
  const float* row = Pout + ((size_t)b * NPATCH + (n - 1)) * Dm;
  const float v0 = row[tid], v1 = row[tid + 256];
  const float tot = block_sum_256(v0 * v0 + v1 * v1, red);
  const float r = rsqrtf(tot * (1.f / Dm) + EPSf);
  Z[(size_t)token * Dm + tid]       = v0 * r * pnw[tid]       + pos[(size_t)n * Dm + tid];
  Z[(size_t)token * Dm + tid + 256] = v1 * r * pnw[tid + 256] + pos[(size_t)n * Dm + tid + 256];
}

__global__ __launch_bounds__(256) void rmsnorm_kernel(
    const float* __restrict__ Z, const float* __restrict__ W, __bf16* __restrict__ Out)
{
  __shared__ float red[8];
  const int t = blockIdx.x, tid = threadIdx.x;
  const float* row = Z + (size_t)t * Dm;
  const float v0 = row[tid], v1 = row[tid + 256];
  const float tot = block_sum_256(v0 * v0 + v1 * v1, red);
  const float r = rsqrtf(tot * (1.f / Dm) + EPSf);
  Out[(size_t)t * Dm + tid]       = f2bf(v0 * r * W[tid]);
  Out[(size_t)t * Dm + tid + 256] = f2bf(v1 * r * W[tid + 256]);
}

// =====================================================================
// Attention: one workgroup per (b, head); 8 waves, one query row per wave.
// =====================================================================
__global__ __launch_bounds__(256) void attention_kernel(
    const float* __restrict__ QKV, float* __restrict__ O)
{
  __shared__ float qs[8][64];
  __shared__ float ps[8][176];
  const int bh = blockIdx.x;
  const int b = bh >> 3, h = bh & 7;
  const int tid = threadIdx.x, lane = tid & 31, wave = tid >> 5;
  const int tbase = b * Ntok;
  const float scale = 0.125f;   // (D/NH)^-0.5
  for (int q0 = 0; q0 < 176; q0 += 8) {
    const int q = q0 + wave;
    const bool act = (q < Ntok);
    if (act) {
      const float* qrow = QKV + (size_t)(tbase + q) * (3 * Dm) + h * 64;
      qs[wave][lane] = qrow[lane];
      qs[wave][lane + 32] = qrow[lane + 32];
    }
    __syncthreads();
    float sc[6];
    float mymax = -1e30f;
    if (act) {
      #pragma unroll
      for (int i = 0; i < 6; ++i) {
        const int j = lane + 32 * i;
        float d = -1e30f;
        if (j < Ntok) {
          const float* krow = QKV + (size_t)(tbase + j) * (3 * Dm) + 512 + h * 64;
          float acc = 0.f;
          #pragma unroll
          for (int dd = 0; dd < 64; ++dd) acc += qs[wave][dd] * krow[dd];
          d = acc * scale;
        }
        sc[i] = d;
        mymax = fmaxf(mymax, d);
      }
    }
    #pragma unroll
    for (int off = 16; off; off >>= 1) mymax = fmaxf(mymax, __shfl_xor(mymax, off, 32));
    float sum = 0.f;
    if (act) {
      #pragma unroll
      for (int i = 0; i < 6; ++i) {
        const int j = lane + 32 * i;
        const float ev = (j < Ntok) ? __expf(sc[i] - mymax) : 0.f;
        sc[i] = ev;
        sum += ev;
      }
    }
    #pragma unroll
    for (int off = 16; off; off >>= 1) sum += __shfl_xor(sum, off, 32);
    if (act) {
      const float inv = 1.f / sum;
      #pragma unroll
      for (int i = 0; i < 6; ++i) {
        const int j = lane + 32 * i;
        if (j < Ntok) ps[wave][j] = sc[i] * inv;
      }
    }
    __syncthreads();
    if (act) {
      float o0 = 0.f, o1 = 0.f;
      for (int j = 0; j < Ntok; ++j) {
        const float pj = ps[wave][j];
        const float* vrow = QKV + (size_t)(tbase + j) * (3 * Dm) + 1024 + h * 64;
        o0 += pj * vrow[lane];
        o1 += pj * vrow[lane + 32];
      }
      float* orow = O + (size_t)(tbase + q) * Dm + h * 64;
      orow[lane] = o0;
      orow[lane + 32] = o1;
    }
    __syncthreads();
  }
}

// =====================================================================
// MoE gating: one wave per token; top-2 routing with atomic slot assignment.
// =====================================================================
__global__ __launch_bounds__(256) void moe_gate_kernel(
    const __bf16* __restrict__ Hn, const float* __restrict__ GW, const float* __restrict__ GB,
    int* __restrict__ counts, int* __restrict__ tok,
    int* __restrict__ e_of, int* __restrict__ s_of, float* __restrict__ w_of)
{
  const int lane = threadIdx.x & 31, wave = threadIdx.x >> 5;
  const int t = blockIdx.x * 8 + wave;
  float logits[8];
  #pragma unroll
  for (int e = 0; e < 8; ++e) {
    float s = 0.f;
    for (int d = lane; d < Dm; d += 32)
      s += bf2f(Hn[(size_t)t * Dm + d]) * GW[e * Dm + d];
    #pragma unroll
    for (int off = 16; off; off >>= 1) s += __shfl_xor(s, off, 32);
    logits[e] = s + GB[e];
  }
  if (lane == 0) {
    int i0 = 0; float m0 = logits[0];
    #pragma unroll
    for (int e = 1; e < 8; ++e) if (logits[e] > m0) { m0 = logits[e]; i0 = e; }
    int i1 = (i0 == 0) ? 1 : 0; float m1 = logits[i1];
    #pragma unroll
    for (int e = 0; e < 8; ++e) if (e != i0 && logits[e] > m1) { m1 = logits[e]; i1 = e; }
    const float e1 = __expf(m1 - m0);
    const float inv = 1.f / (1.f + e1);
    const int s0 = atomicAdd(&counts[i0], 1);
    const int s1 = atomicAdd(&counts[i1], 1);
    tok[i0 * Ttok + s0] = t;
    tok[i1 * Ttok + s1] = t;
    e_of[2 * t] = i0; e_of[2 * t + 1] = i1;
    s_of[2 * t] = s0; s_of[2 * t + 1] = s1;
    w_of[2 * t] = inv; w_of[2 * t + 1] = e1 * inv;
  }
}

// deterministic per-token combine (no float atomics)
__global__ __launch_bounds__(256) void moe_combine_kernel(
    const float* __restrict__ Y, const int* __restrict__ e_of, const int* __restrict__ s_of,
    const float* __restrict__ w_of, float* __restrict__ Z)
{
  const int t = blockIdx.x, tid = threadIdx.x;
  const int e0 = e_of[2 * t], e1 = e_of[2 * t + 1];
  const int s0 = s_of[2 * t], s1 = s_of[2 * t + 1];
  const float w0 = w_of[2 * t], w1 = w_of[2 * t + 1];
  const float* y0 = Y + ((size_t)e0 * Ttok + s0) * Dm;
  const float* y1 = Y + ((size_t)e1 * Ttok + s1) * Dm;
  float* zr = Z + (size_t)t * Dm;
  zr[tid]       += w0 * y0[tid]       + w1 * y1[tid];
  zr[tid + 256] += w0 * y0[tid + 256] + w1 * y1[tid + 256];
}

// final norm of cls token + classifier head
__global__ __launch_bounds__(256) void head_kernel(
    const float* __restrict__ Z, const float* __restrict__ NW,
    const float* __restrict__ HW, const float* __restrict__ HB, float* __restrict__ out)
{
  __shared__ float nz[512];
  __shared__ float red[8];
  const int b = blockIdx.x, tid = threadIdx.x;
  const int lane = tid & 31, wave = tid >> 5;
  const float* row = Z + (size_t)(b * Ntok) * Dm;
  const float v0 = row[tid], v1 = row[tid + 256];
  const float tot = block_sum_256(v0 * v0 + v1 * v1, red);
  const float r = rsqrtf(tot * (1.f / Dm) + EPSf);
  nz[tid] = v0 * r * NW[tid];
  nz[tid + 256] = v1 * r * NW[tid + 256];
  __syncthreads();
  for (int c = wave; c < NCLS; c += 8) {
    float s = 0.f;
    for (int d = lane; d < Dm; d += 32) s += nz[d] * HW[(size_t)c * Dm + d];
    #pragma unroll
    for (int off = 16; off; off >>= 1) s += __shfl_xor(s, off, 32);
    if (lane == 0) out[b * NCLS + c] = s + HB[c];
  }
}

// weight layout helpers
__global__ __launch_bounds__(256) void transpose_convert_kernel(
    const float* __restrict__ in, __bf16* __restrict__ out, int R, int Cc)
{
  const size_t i = (size_t)blockIdx.x * 256 + threadIdx.x;
  if (i >= (size_t)R * Cc) return;
  const int r = (int)(i / Cc), c = (int)(i % Cc);
  out[(size_t)c * R + r] = f2bf(in[i]);
}

__global__ __launch_bounds__(256) void convert_bf16_kernel(
    const float* __restrict__ in, __bf16* __restrict__ out, size_t n)
{
  const size_t i = (size_t)blockIdx.x * 256 + threadIdx.x;
  if (i < n) out[i] = f2bf(in[i]);
}

// =====================================================================
extern "C" void kernel_launch(void* const* d_in, const int* in_sizes, int n_in,
                              void* d_out, int out_size, void* d_ws, size_t ws_size,
                              hipStream_t stream)
{
  (void)in_sizes; (void)n_in; (void)out_size; (void)ws_size;
  const float* x            = (const float*)d_in[0];
  const float* conv_w       = (const float*)d_in[1];
  const float* conv_b       = (const float*)d_in[2];
  const float* patch_norm_w = (const float*)d_in[3];
  const float* cls_token    = (const float*)d_in[4];
  const float* pos_embed    = (const float*)d_in[5];
  const float* norm1_w      = (const float*)d_in[6];
  const float* qkv_w        = (const float*)d_in[7];
  const float* qkv_b        = (const float*)d_in[8];
  const float* proj_w       = (const float*)d_in[9];
  const float* proj_b       = (const float*)d_in[10];
  const float* norm2_w      = (const float*)d_in[11];
  const float* gate_w       = (const float*)d_in[12];
  const float* gate_b       = (const float*)d_in[13];
  const float* w1           = (const float*)d_in[14];
  const float* b1           = (const float*)d_in[15];
  const float* w2           = (const float*)d_in[16];
  const float* b2           = (const float*)d_in[17];
  const float* norm_w       = (const float*)d_in[18];
  const float* head_w       = (const float*)d_in[19];
  const float* head_b       = (const float*)d_in[20];
  float* out = (float*)d_out;

  char* p = (char*)d_ws;
  auto carve = [&](size_t bytes) -> void* {
    char* r = p;
    p += (bytes + 255) & ~(size_t)255;
    return (void*)r;
  };
  float*  Z     = (float*) carve((size_t)Ttok * Dm * 4);
  __bf16* Hn    = (__bf16*)carve((size_t)Ttok * Dm * 2);
  float*  QKV   = (float*) carve((size_t)Ttok * 3 * Dm * 4);
  float*  Obuf  = (float*) carve((size_t)Ttok * Dm * 4);
  __bf16* Obf   = (__bf16*)carve((size_t)Ttok * Dm * 2);
  __bf16* Acol  = (__bf16*)carve((size_t)TP * KPATCH * 2);
  __bf16* Wc    = (__bf16*)carve((size_t)KPATCH * Dm * 2);
  float*  Pout  = (float*) carve((size_t)TP * Dm * 4);
  __bf16* Wqkv  = (__bf16*)carve((size_t)Dm * 3 * Dm * 2);
  __bf16* Wproj = (__bf16*)carve((size_t)Dm * Dm * 2);
  __bf16* W1b   = (__bf16*)carve((size_t)Ee * Dm * HIDn * 2);
  __bf16* W2b   = (__bf16*)carve((size_t)Ee * HIDn * Dm * 2);
  __bf16* Hid   = (__bf16*)carve((size_t)Ee * Ttok * HIDn * 2);
  float*  Ybuf  = (float*) carve((size_t)Ee * Ttok * Dm * 4);
  int*    counts= (int*)   carve((size_t)Ee * 4);
  int*    tok   = (int*)   carve((size_t)Ee * Ttok * 4);
  int*    e_of  = (int*)   carve((size_t)Ttok * 2 * 4);
  int*    s_of  = (int*)   carve((size_t)Ttok * 2 * 4);
  float*  w_of  = (float*) carve((size_t)Ttok * 2 * 4);

  // ---- patch embed (im2col + WMMA GEMM) ----
  {
    const size_t ne = (size_t)TP * KPATCH;
    im2col_kernel<<<(unsigned)((ne + 255) / 256), 256, 0, stream>>>(x, Acol);
    const size_t nw = (size_t)Dm * KPATCH;
    transpose_convert_kernel<<<(unsigned)((nw + 255) / 256), 256, 0, stream>>>(conv_w, Wc, Dm, KPATCH);
    dim3 g((TP + 63) / 64, Dm / 64);
    gemm_bf16_kernel<<<g, 256, 0, stream>>>(Acol, Wc, conv_b, Pout, TP, Dm, KPATCH, 0);
    patch_post_kernel<<<Ttok, 256, 0, stream>>>(Pout, patch_norm_w, cls_token, pos_embed, Z);
  }

  for (int l = 0; l < Ll; ++l) {
    // attention block
    rmsnorm_kernel<<<Ttok, 256, 0, stream>>>(Z, norm1_w + (size_t)l * Dm, Hn);
    {
      const size_t nw = (size_t)3 * Dm * Dm;
      transpose_convert_kernel<<<(unsigned)((nw + 255) / 256), 256, 0, stream>>>(
          qkv_w + (size_t)l * 3 * Dm * Dm, Wqkv, 3 * Dm, Dm);
      dim3 g((Ttok + 63) / 64, (3 * Dm) / 64);
      gemm_bf16_kernel<<<g, 256, 0, stream>>>(Hn, Wqkv, qkv_b + (size_t)l * 3 * Dm, QKV,
                                              Ttok, 3 * Dm, Dm, 0);
    }
    attention_kernel<<<Bsz * NHh, 256, 0, stream>>>(QKV, Obuf);
    {
      const size_t ne = (size_t)Ttok * Dm;
      convert_bf16_kernel<<<(unsigned)((ne + 255) / 256), 256, 0, stream>>>(Obuf, Obf, ne);
      const size_t nw = (size_t)Dm * Dm;
      transpose_convert_kernel<<<(unsigned)((nw + 255) / 256), 256, 0, stream>>>(
          proj_w + (size_t)l * Dm * Dm, Wproj, Dm, Dm);
      dim3 g((Ttok + 63) / 64, Dm / 64);
      gemm_bf16_kernel<<<g, 256, 0, stream>>>(Obf, Wproj, proj_b + (size_t)l * Dm, Z,
                                              Ttok, Dm, Dm, 1);
    }
    // MoE block (top-2 only)
    rmsnorm_kernel<<<Ttok, 256, 0, stream>>>(Z, norm2_w + (size_t)l * Dm, Hn);
    hipMemsetAsync(counts, 0, Ee * sizeof(int), stream);
    moe_gate_kernel<<<Ttok / 8, 256, 0, stream>>>(Hn, gate_w + (size_t)l * Ee * Dm,
                                                  gate_b + (size_t)l * Ee,
                                                  counts, tok, e_of, s_of, w_of);
    {
      const size_t nw = (size_t)Ee * Dm * HIDn;
      convert_bf16_kernel<<<(unsigned)((nw + 255) / 256), 256, 0, stream>>>(w1 + (size_t)l * nw, W1b, nw);
      convert_bf16_kernel<<<(unsigned)((nw + 255) / 256), 256, 0, stream>>>(w2 + (size_t)l * nw, W2b, nw);
    }
    {
      dim3 g1((Ttok + 63) / 64, HIDn / 64, Ee);
      moe_gemm1_kernel<<<g1, 256, 0, stream>>>(Hn, W1b, b1 + (size_t)l * Ee * HIDn, counts, tok, Hid);
      dim3 g2((Ttok + 63) / 64, Dm / 64, Ee);
      moe_gemm2_kernel<<<g2, 256, 0, stream>>>(Hid, W2b, b2 + (size_t)l * Ee * Dm, counts, Ybuf);
      moe_combine_kernel<<<Ttok, 256, 0, stream>>>(Ybuf, e_of, s_of, w_of, Z);
    }
  }

  head_kernel<<<Bsz, 256, 0, stream>>>(Z, norm_w, head_w, head_b, out);
}
